// VQVAE_7679401525805
// MI455X (gfx1250) — matrix operand order repaired and checked by reading, not compiled, not run on hardware
//
#include <hip/hip_runtime.h>

// ---------------------------------------------------------------------------
// VQ-VAE forward for MI455X (gfx1250, wave32, WMMA).
// 8 GEMMs via v_wmma_f32_16x16x32_bf16 (bf16 in, f32 accum).
// Weights pre-transposed to bf16 [N,K] once per launch when workspace allows
// (then A and B tiles are identical contiguous copies, eligible for
// GLOBAL_LOAD_ASYNC_TO_LDS_B128); otherwise f32 weights are converted/
// transposed on the fly in the loader.
// ---------------------------------------------------------------------------

typedef __bf16 bf16_t;
typedef __attribute__((ext_vector_type(16))) __bf16 v16bf;
typedef __attribute__((ext_vector_type(8)))  __bf16 v8bf;
typedef __attribute__((ext_vector_type(8)))  float   v8f;
typedef __attribute__((ext_vector_type(4)))  float   v4f;
typedef __attribute__((ext_vector_type(4)))  int     v4i;

#define TILE_M 128
#define TILE_N 128
#define TILE_K 32
#define LDS_STRIDE 40   // 32 + 8 pad (bf16 elems); 80B row stride, 16B aligned

#if __has_builtin(__builtin_amdgcn_global_load_async_to_lds_b128)
#define HAVE_ASYNC_LDS 1
typedef __attribute__((address_space(1))) v4i glb_v4i;   // global int4
typedef __attribute__((address_space(3))) v4i lds_v4i;   // LDS int4
#else
#define HAVE_ASYNC_LDS 0
#endif

union FragAB { v16bf v; v8bf h[2]; };

// 16-byte global->LDS copy: async (ASYNCcnt, no VGPR round trip) when available.
__device__ __forceinline__ void copy16_g2l(const bf16_t* __restrict__ g, bf16_t* l)
{
#if HAVE_ASYNC_LDS
    __builtin_amdgcn_global_load_async_to_lds_b128((glb_v4i*)g, (lds_v4i*)l, 0, 0);
#else
    *(v8bf*)l = *(const v8bf*)g;
#endif
}

__device__ __forceinline__ void wait_async_copies()
{
#if HAVE_ASYNC_LDS
#if __has_builtin(__builtin_amdgcn_s_wait_asynccnt)
    __builtin_amdgcn_s_wait_asynccnt(0);
#else
    asm volatile("s_wait_asynccnt 0x0" ::: "memory");
#endif
#endif
}

__global__ __launch_bounds__(256)
void f32_to_bf16_kernel(const float* __restrict__ in, bf16_t* __restrict__ out, int n)
{
    int i = blockIdx.x * blockDim.x + threadIdx.x;
    if (i < n) out[i] = (bf16_t)in[i];
}

// W [K,N] f32 -> WT [N,K] bf16 (LDS-tiled 32x32 for coalescing on both sides)
__global__ __launch_bounds__(256)
void transpose_f32_to_bf16T(const float* __restrict__ W, bf16_t* __restrict__ WT,
                            int K, int N)
{
    __shared__ float tile[32][33];
    const int bn = blockIdx.x * 32;           // n
    const int bk = blockIdx.y * 32;           // k
    const int tx = threadIdx.x & 31;
    const int ty = threadIdx.x >> 5;          // 0..7
#pragma unroll
    for (int i = 0; i < 32; i += 8)
        tile[ty + i][tx] = W[(size_t)(bk + ty + i) * N + bn + tx];
    __syncthreads();
#pragma unroll
    for (int i = 0; i < 32; i += 8)
        WT[(size_t)(bn + ty + i) * K + bk + tx] = (bf16_t)tile[tx][ty + i];
}

// C = act(A[M,Kd](bf16) @ B + bias)
//   BT=1: Bmat = WT [N,Kd] bf16 row-major (pre-transposed)
//   BT=0: Bmat = W  [Kd,N] f32 row-major (converted/transposed on the fly)
template<int RELU, int WRITE_F32, int WRITE_BF16, int BT>
__global__ __launch_bounds__(256)
void gemm_bf16_wmma(const bf16_t* __restrict__ A,
                    const void*   __restrict__ Bmat,
                    const float*  __restrict__ bias,
                    bf16_t* __restrict__ outB,
                    float*  __restrict__ outF,
                    int Kd, int N)
{
    __shared__ bf16_t sA[TILE_M][LDS_STRIDE];   // [m][k]
    __shared__ bf16_t sB[TILE_N][LDS_STRIDE];   // B^T tile: [n][k]

    const int tid  = threadIdx.x;
    const int lane = tid & 31;
    const int wave = tid >> 5;       // 0..7
    const int wm   = wave >> 1;      // 0..3 -> 32 rows each
    const int wn   = wave & 1;       // 0..1 -> 64 cols each
    const int half = lane >> 4;      // 0/1
    const int l16  = lane & 15;

    const int block_m = blockIdx.y * TILE_M;
    const int block_n = blockIdx.x * TILE_N;

    v8f acc[2][4];
#pragma unroll
    for (int mt = 0; mt < 2; ++mt)
#pragma unroll
        for (int nt = 0; nt < 4; ++nt)
            acc[mt][nt] = (v8f){0.f,0.f,0.f,0.f,0.f,0.f,0.f,0.f};

    // per-thread cooperative load coordinates (16B granularity)
    const int a_row = tid >> 2;            // 0..63, two passes
    const int a_col = (tid & 3) * 8;       // 0,8,16,24 (bf16 elems)
    const int b_k   = tid >> 5;            // 0..7, four passes  (BT=0 path)
    const int b_n   = (tid & 31) * 4;      // 0..124             (BT=0 path)

    for (int k0 = 0; k0 < Kd; k0 += TILE_K) {
        // ---- stage A tile ----
#pragma unroll
        for (int p = 0; p < 2; ++p) {
            int r = p * 64 + a_row;
            copy16_g2l(A + (size_t)(block_m + r) * Kd + k0 + a_col, &sA[r][a_col]);
        }
        // ---- stage B tile ----
        if (BT) {
            const bf16_t* WT = (const bf16_t*)Bmat;   // [N, Kd]
#pragma unroll
            for (int p = 0; p < 2; ++p) {
                int r = p * 64 + a_row;               // n within tile
                copy16_g2l(WT + (size_t)(block_n + r) * Kd + k0 + a_col, &sB[r][a_col]);
            }
        } else {
            const float* W = (const float*)Bmat;      // [Kd, N]
#pragma unroll
            for (int p = 0; p < 4; ++p) {
                int kk = p * 8 + b_k;
                v4f w4 = *(const v4f*)(W + (size_t)(k0 + kk) * N + block_n + b_n);
                sB[b_n + 0][kk] = (bf16_t)w4.x;
                sB[b_n + 1][kk] = (bf16_t)w4.y;
                sB[b_n + 2][kk] = (bf16_t)w4.z;
                sB[b_n + 3][kk] = (bf16_t)w4.w;
            }
        }
        // prefetch next k-step's A tile into cache while we compute
        if (k0 + TILE_K < Kd)
            __builtin_prefetch(A + (size_t)(block_m + a_row) * Kd + k0 + TILE_K + a_col, 0, 1);

        wait_async_copies();
        __syncthreads();

        // ---- fragments per ISA 16-bit A / B VGPR layouts (wave32) ----
        FragAB a[2], b[4];
#pragma unroll
        for (int mt = 0; mt < 2; ++mt) {
            int row = wm * 32 + mt * 16 + l16;
            a[mt].h[0] = *(const v8bf*)&sA[row][half * 8];        // K = half*8 .. +7
            a[mt].h[1] = *(const v8bf*)&sA[row][16 + half * 8];   // K = 16+half*8 .. +7
        }
#pragma unroll
        for (int nt = 0; nt < 4; ++nt) {
            int col = wn * 64 + nt * 16 + l16;
            b[nt].h[0] = *(const v8bf*)&sB[col][half * 8];        // K = half*8 .. +7
            b[nt].h[1] = *(const v8bf*)&sB[col][16 + half * 8];   // K = 16+half*8 .. +7
        }

#pragma unroll
        for (int mt = 0; mt < 2; ++mt)
#pragma unroll
            for (int nt = 0; nt < 4; ++nt)
                acc[mt][nt] = __builtin_amdgcn_wmma_f32_16x16x32_bf16(
                    false, a[mt].v, false, b[nt].v,
                    (short)0, acc[mt][nt], false, false);

        __syncthreads();
    }

    // ---- epilogue: C/D layout: VGPR i -> M = half*8 + i, N = l16 ----
#pragma unroll
    for (int nt = 0; nt < 4; ++nt) {
        int colg = block_n + wn * 64 + nt * 16 + l16;
        float bv = bias[colg];
#pragma unroll
        for (int mt = 0; mt < 2; ++mt) {
            int rbase = block_m + wm * 32 + mt * 16 + half * 8;
#pragma unroll
            for (int i = 0; i < 8; ++i) {
                float v = acc[mt][nt][i] + bv;
                if (RELU) v = v > 0.f ? v : 0.f;
                size_t off = (size_t)(rbase + i) * N + colg;
                if (WRITE_F32)  outF[off] = v;
                if (WRITE_BF16) outB[off] = (bf16_t)v;
            }
        }
    }
}

// One thread per latent vector: argmin over 512 codes of ||z - c||^2.
__global__ __launch_bounds__(256)
void vq_kernel(const float* __restrict__ z_e,      // [131072, 64]
               const float* __restrict__ codebook, // [512, 64]
               float* __restrict__ zq_st,
               float* __restrict__ zq,
               float* __restrict__ idx_out,        // [131072] (float-encoded int)
               bf16_t* __restrict__ zq_bf)         // decoder input
{
    int t = blockIdx.x * 256 + threadIdx.x;
    const float* v = z_e + (size_t)t * 64;

    float f[64];
#pragma unroll
    for (int j = 0; j < 64; j += 4) {
        v4f x = *(const v4f*)(v + j);
        f[j + 0] = x.x; f[j + 1] = x.y; f[j + 2] = x.z; f[j + 3] = x.w;
    }

    float best = 3.4e38f;
    int bi = 0;
    for (int k = 0; k < 512; ++k) {
        const float* c = codebook + (size_t)k * 64;   // wave-uniform -> scalar path
        float d = 0.f;
#pragma unroll
        for (int j = 0; j < 64; ++j) {
            float df = f[j] - c[j];
            d = fmaf(df, df, d);
        }
        if (d < best) { best = d; bi = k; }           // '<' keeps first min (argmin)
    }

    idx_out[t] = (float)bi;
    const float* c = codebook + (size_t)bi * 64;
    size_t base = (size_t)t * 64;
#pragma unroll
    for (int j = 0; j < 64; ++j) {
        float w = c[j];
        zq_st[base + j] = w;
        zq[base + j]    = w;
        zq_bf[base + j] = (bf16_t)w;
    }
}

extern "C" void kernel_launch(void* const* d_in, const int* in_sizes, int n_in,
                              void* d_out, int out_size, void* d_ws, size_t ws_size,
                              hipStream_t stream)
{
    (void)in_sizes; (void)n_in; (void)out_size;

    const int B = 4096, NB_LINKS = 1024, H = 2048, ZF = 2048, NBZ = 32;

    const float* x = (const float*)d_in[0];
    const float *ew[4], *eb[4], *dw[4], *db[4];
    for (int i = 0; i < 4; ++i) {
        ew[i] = (const float*)d_in[1 + i];
        eb[i] = (const float*)d_in[5 + i];
        dw[i] = (const float*)d_in[9 + i];
        db[i] = (const float*)d_in[13 + i];
    }
    const float* codebook = (const float*)d_in[17];

    float* out   = (float*)d_out;
    float* z_e   = out;                                             // [B, 2048]
    float* zq_st = out + (size_t)B * ZF;                            // [B, 32, 64]
    float* zq    = out + (size_t)2 * B * ZF;                        // [B, 32, 64]
    float* xhat  = out + (size_t)3 * B * ZF;                        // [B, 1024]
    float* idxf  = out + (size_t)3 * B * ZF + (size_t)B * NB_LINKS; // [B, 32]

    bf16_t* xb   = (bf16_t*)d_ws;               // [B, 1024] bf16
    bf16_t* bufA = xb   + (size_t)B * NB_LINKS; // [B, 2048] bf16
    bf16_t* bufB = bufA + (size_t)B * H;        // [B, 2048] bf16

    // optional pre-transposed bf16 weights (58.7 MB) if workspace allows
    const size_t wKN[8][2] = { {NB_LINKS,H},{H,H},{H,H},{H,ZF},
                               {ZF,H},{H,H},{H,H},{H,NB_LINKS} };
    size_t base_elems = (size_t)B * NB_LINKS + 2 * (size_t)B * H;
    size_t wt_elems = 0;
    for (int i = 0; i < 8; ++i) wt_elems += wKN[i][0] * wKN[i][1];
    const bool useWT = ws_size >= (base_elems + wt_elems) * sizeof(bf16_t);

    bf16_t* wt[8];
    if (useWT) {
        bf16_t* p = bufB + (size_t)B * H;
        for (int i = 0; i < 8; ++i) { wt[i] = p; p += wKN[i][0] * wKN[i][1]; }
    }

    dim3 blk(256);

    // x -> bf16
    {
        int n = B * NB_LINKS;
        f32_to_bf16_kernel<<<dim3((n + 255) / 256), blk, 0, stream>>>(x, xb, n);
    }

    const float* wsrc[8] = { ew[0], ew[1], ew[2], ew[3], dw[0], dw[1], dw[2], dw[3] };
    if (useWT) {
        for (int i = 0; i < 8; ++i) {
            int K = (int)wKN[i][0], N = (int)wKN[i][1];
            transpose_f32_to_bf16T<<<dim3(N / 32, K / 32), blk, 0, stream>>>(wsrc[i], wt[i], K, N);
        }
    }

    dim3 gH(H / TILE_N, B / TILE_M);          // 16 x 32
    dim3 gO(NB_LINKS / TILE_N, B / TILE_M);   // 8 x 32

#define GEMM(RELU, WF, WB, Ain, Widx, bia, oB, oF, Kd, Nn, grid)                          \
    do {                                                                                  \
        if (useWT) gemm_bf16_wmma<RELU,WF,WB,1><<<grid, blk, 0, stream>>>(                \
                        (Ain), (const void*)wt[Widx], (bia), (oB), (oF), (Kd), (Nn));     \
        else       gemm_bf16_wmma<RELU,WF,WB,0><<<grid, blk, 0, stream>>>(                \
                        (Ain), (const void*)wsrc[Widx], (bia), (oB), (oF), (Kd), (Nn));   \
    } while (0)

    // encoder: 1024 -> 2048 -> 2048 -> 2048 -> 2048 (last linear, f32 out = z_e)
    GEMM(1,0,1, xb,   0, eb[0], bufA, nullptr, NB_LINKS, H, gH);
    GEMM(1,0,1, bufA, 1, eb[1], bufB, nullptr, H, H, gH);
    GEMM(1,0,1, bufB, 2, eb[2], bufA, nullptr, H, H, gH);
    GEMM(0,1,0, bufA, 3, eb[3], nullptr, z_e, H, ZF, gH);

    // vector quantization (also produces bf16 decoder input in bufA)
    vq_kernel<<<dim3(B * NBZ / 256), blk, 0, stream>>>(z_e, codebook, zq_st, zq, idxf, bufA);

    // decoder: 2048 -> 2048 -> 2048 -> 2048 -> 1024 (last linear, f32 out = x_hat)
    GEMM(1,0,1, bufA, 4, db[0], bufB, nullptr, ZF, H, gH);
    GEMM(1,0,1, bufB, 5, db[1], bufA, nullptr, H, H, gH);
    GEMM(1,0,1, bufA, 6, db[2], bufB, nullptr, H, H, gH);
    GEMM(0,1,0, bufB, 7, db[3], nullptr, xhat, H, NB_LINKS, gO);
#undef GEMM
}
